// GRUAttentionDecoder_82274393522215
// MI455X (gfx1250) — matrix operand order repaired and checked by reading, not compiled, not run on hardware
//
#include <hip/hip_runtime.h>
#include <cstdint>
#include <cstddef>

// ---------------- types ----------------
typedef __attribute__((ext_vector_type(16))) __bf16       v16bf;
typedef __attribute__((ext_vector_type(8)))  float        v8f;
typedef __attribute__((ext_vector_type(4)))  unsigned int u32x4;
typedef __attribute__((ext_vector_type(4)))  unsigned int v4u;
typedef __attribute__((ext_vector_type(8)))  int          v8i;
typedef __attribute__((ext_vector_type(4)))  int          v4i;

union Frag { u32x4 q[2]; v16bf v; };

#define LDSS 40   // padded LDS row stride in halves (80B = 64B row + 16B pad)

// problem constants
#define BB   256
#define SS   100
#define HH   1024
#define H3   3072
#define FF   498
#define BN   40
#define E4   2048
#define THIST 10
#define DEC  30

#if defined(__gfx1250__) && __has_builtin(__builtin_amdgcn_tensor_load_to_lds) && \
    __has_builtin(__builtin_amdgcn_s_wait_tensorcnt)
#define USE_TDM 1
#else
#define USE_TDM 0
#endif

__device__ __forceinline__ unsigned short f32_to_bf16(float x) {
    union { float f; unsigned int u; } v; v.f = x;
    unsigned int u = v.u;
    if ((u & 0x7F800000u) != 0x7F800000u)      // normal/denorm: round-nearest-even
        u = u + 0x7FFFu + ((u >> 16) & 1u);
    return (unsigned short)(u >> 16);
}

#if USE_TDM
// Issue a TDM load of a 2D bf16 tile: tile_dim0=32 halves (64B) x tile_dim1=128 rows,
// row stride lda (halves) in global, into LDS at lds_byte_addr with 16B padding per
// 64B row (pad_interval code 3 = 16 DWORDs, pad_amount code 3 = 4 DWORDs) -> LDSS=40.
// 2D tensor: descriptor groups 2/3 (and trailing group operand) are zero.
__device__ __forceinline__ void tdm_load_tile(const unsigned short* gptr,
                                              unsigned lds_byte_addr,
                                              unsigned lda_halves) {
    unsigned long long ga = (unsigned long long)(size_t)gptr;
    const unsigned td0 = 0x40000000u;   // huge tensor dims: no OOB clipping
    const unsigned td1 = 0x40000000u;
    v4u g0;
    g0[0] = 1u;                                   // count=1, user descriptor
    g0[1] = lds_byte_addr;                        // D#.lds_addr
    g0[2] = (unsigned)ga;                         // global_addr[31:0]
    g0[3] = (unsigned)(ga >> 32) | (2u << 30);    // global_addr[56:32] | type=2
    v8i g1;
    g1[0] = (int)((1u << 16)      // data_size = 2 bytes
                | (1u << 20)      // pad_enable
                | (3u << 22)      // pad_interval: 16 DWORDs
                | (3u << 25));    // pad_amount: 4 DWORDs
    g1[1] = (int)((td0 & 0xFFFFu) << 16);                             // dim0[15:0]
    g1[2] = (int)(((td0 >> 16) & 0xFFFFu) | ((td1 & 0xFFFFu) << 16)); // dim0 hi | dim1 lo
    g1[3] = (int)(((td1 >> 16) & 0xFFFFu) | (32u << 16));             // dim1 hi | tile_dim0=32
    g1[4] = (int)128u;                                                // tile_dim1=128, tile_dim2=0
    g1[5] = (int)lda_halves;                                          // tensor_dim0_stride[31:0]
    g1[6] = 0;                                                        // stride hi | dim1_stride lo
    g1[7] = 0;
    v4i z4 = {0, 0, 0, 0};
    v8i z8 = {0, 0, 0, 0, 0, 0, 0, 0};
    __builtin_amdgcn_tensor_load_to_lds(g0, g1, z4, z4, z8, 0);
}
#endif

// ---------------- WMMA GEMM: C[M,N] = A[M,K](bf16) @ Bt[N,K]^T(bf16) + bias ----------------
// A row-major with leading dim lda; Bt is B transposed, row-major N x K (ld = K).
// M must be a multiple of 128; K a multiple of 32; Bt must have ceil(N/128)*128 valid rows.
__global__ __launch_bounds__(256)
void gemm_bf16_kernel(const unsigned short* __restrict__ A, int lda,
                      const unsigned short* __restrict__ Bt,
                      const float* __restrict__ bias,
                      float* __restrict__ C, int ldc,
                      int M, int N_store, int K)
{
    __shared__ __attribute__((aligned(16))) unsigned short As[2][128 * LDSS];
    __shared__ __attribute__((aligned(16))) unsigned short Bs[2][128 * LDSS];

    const int tid  = threadIdx.x;
    const int lane = tid & 31;
    const int wave = tid >> 5;
    const int bm   = blockIdx.x * 128;
    const int bn   = blockIdx.y * 128;
    const int wm   = (wave & 3) * 32;   // wave's M offset in tile
    const int wn   = (wave >> 2) * 64;  // wave's N offset in tile
    const int lrow = lane & 15;
    const int akoff = (lane < 16) ? 0 : 8;   // A frag K sub-offset (halves)
    const int bkoff = (lane < 16) ? 0 : 8;   // B frag K sub-offset (halves)

    v8f acc[2][4];
#pragma unroll
    for (int i = 0; i < 2; ++i)
#pragma unroll
        for (int j = 0; j < 4; ++j)
#pragma unroll
            for (int r = 0; r < 8; ++r) acc[i][j][r] = 0.0f;

    const int nsteps = K >> 5;

#if USE_TDM
    const unsigned short* Abase = A  + (size_t)bm * lda;
    const unsigned short* Bbase = Bt + (size_t)bn * K;
    if (wave == 0) {
        tdm_load_tile(Abase, (unsigned)(size_t)&As[0][0], (unsigned)lda);
        tdm_load_tile(Bbase, (unsigned)(size_t)&Bs[0][0], (unsigned)K);
    }
#endif

    for (int i = 0; i < nsteps; ++i) {
        const int cur = i & 1;
#if USE_TDM
        if (wave == 0) {
            if (i + 1 < nsteps) {
                // queue next tile into the other buffer, then wait for current pair
                tdm_load_tile(Abase + (size_t)(i + 1) * 32,
                              (unsigned)(size_t)&As[cur ^ 1][0], (unsigned)lda);
                tdm_load_tile(Bbase + (size_t)(i + 1) * 32,
                              (unsigned)(size_t)&Bs[cur ^ 1][0], (unsigned)K);
                __builtin_amdgcn_s_wait_tensorcnt(2);
            } else {
                __builtin_amdgcn_s_wait_tensorcnt(0);
            }
        }
        __syncthreads();   // current buffer's TDM data now visible to all waves
#else
        // fallback: cooperative VGPR staging, single-buffered
        const int k0 = i * 32;
#pragma unroll
        for (int p = 0; p < 2; ++p) {
            int idx = p * 256 + tid;
            int r   = idx >> 2;
            int cg  = (idx & 3) * 8;
            u32x4 av = *(const u32x4*)(A  + (size_t)(bm + r) * lda + k0 + cg);
            *(u32x4*)(&As[cur][r * LDSS + cg]) = av;
            u32x4 bv = *(const u32x4*)(Bt + (size_t)(bn + r) * K   + k0 + cg);
            *(u32x4*)(&Bs[cur][r * LDSS + cg]) = bv;
        }
        __syncthreads();
#endif

        v16bf afrag[2];
#pragma unroll
        for (int ms = 0; ms < 2; ++ms) {
            const unsigned short* src = &As[cur][(wm + ms * 16 + lrow) * LDSS + akoff];
            Frag f;
            f.q[0] = *(const u32x4*)(src);
            f.q[1] = *(const u32x4*)(src + 16);
            afrag[ms] = f.v;
        }
        v16bf bfrag[4];
#pragma unroll
        for (int ns = 0; ns < 4; ++ns) {
            const unsigned short* src = &Bs[cur][(wn + ns * 16 + lrow) * LDSS + bkoff];
            Frag f;
            f.q[0] = *(const u32x4*)(src);
            f.q[1] = *(const u32x4*)(src + 16);
            bfrag[ns] = f.v;
        }
#pragma unroll
        for (int ms = 0; ms < 2; ++ms)
#pragma unroll
            for (int ns = 0; ns < 4; ++ns)
                acc[ms][ns] = __builtin_amdgcn_wmma_f32_16x16x32_bf16(
                    false, afrag[ms], false, bfrag[ns],
                    (short)0, acc[ms][ns], false, false);
        __syncthreads();   // all waves done reading buffer `cur`; safe to overwrite next iter
    }

    // store: VGPR r -> row r (lanes 0-15) / row r+8 (lanes 16-31), col = lane&15
    const int rsub = (lane < 16) ? 0 : 8;
#pragma unroll
    for (int ms = 0; ms < 2; ++ms) {
#pragma unroll
        for (int ns = 0; ns < 4; ++ns) {
            int col = bn + wn + ns * 16 + lrow;
            if (col < N_store) {
                float bv = bias ? bias[col] : 0.0f;
                int rbase = bm + wm + ms * 16 + rsub;
#pragma unroll
                for (int r = 0; r < 8; ++r)
                    C[(size_t)(rbase + r) * ldc + col] = acc[ms][ns][r] + bv;
            }
        }
    }
}

// ---------------- weight convert + transpose: Wt[n,k] = bf16(W[k,n]), zero-pad n>=N ----------------
__global__ void convert_transpose_kernel(const float* __restrict__ W,
                                         unsigned short* __restrict__ Wt,
                                         int K, int N, int Npad)
{
    int idx = blockIdx.x * blockDim.x + threadIdx.x;
    if (idx >= Npad * K) return;
    int n = idx / K, k = idx % K;
    float v = (n < N) ? W[(size_t)k * N + n] : 0.0f;
    Wt[idx] = f32_to_bf16(v);
}

// plain f32 -> bf16
__global__ void convert_kernel(const float* __restrict__ x,
                               unsigned short* __restrict__ y, int n)
{
    int idx = blockIdx.x * blockDim.x + threadIdx.x;
    if (idx < n) y[idx] = f32_to_bf16(x[idx]);
}

// h0 GEMM input gather: y[b, l*512+e] = bf16(eh[l,b,e])
__global__ void h0_gather_kernel(const float* __restrict__ eh,
                                 unsigned short* __restrict__ y)
{
    int idx = blockIdx.x * blockDim.x + threadIdx.x;
    if (idx >= BB * E4) return;
    int b = idx / E4, c = idx % E4;
    int l = c / 512, e = c % 512;
    y[idx] = f32_to_bf16(eh[((size_t)l * BB + b) * 512 + e]);
}

// f32 h (B x H) -> bf16 into concat buffer first half (row stride 2048)
__global__ void h_to_concat_kernel(const float* __restrict__ h,
                                   unsigned short* __restrict__ concat)
{
    int idx = blockIdx.x * blockDim.x + threadIdx.x;
    if (idx >= BB * HH) return;
    int b = idx >> 10, j = idx & (HH - 1);
    concat[(size_t)b * E4 + j] = f32_to_bf16(h[idx]);
}

// ---------------- bottleneck (plain f32; tiny GEMMs with awkward K=498/N=40) ----------------
__global__ void bottleneck1_kernel(const float* __restrict__ x,
                                   int rows_per_b, int b_stride, int row_stride,
                                   const float* __restrict__ W1,
                                   const float* __restrict__ b1,
                                   float* __restrict__ t, int M)
{
    int idx = blockIdx.x * blockDim.x + threadIdx.x;
    if (idx >= M * BN) return;
    int m = idx / BN, j = idx % BN;
    const float* xr = x + (size_t)(m / rows_per_b) * b_stride
                        + (size_t)(m % rows_per_b) * row_stride;
    float acc = b1[j];
    for (int k = 0; k < FF; ++k) acc += xr[k] * W1[k * BN + j];
    t[idx] = fmaxf(acc, 0.0f);
}

__global__ void bottleneck2_kernel(const float* __restrict__ t,
                                   const float* __restrict__ W2,
                                   const float* __restrict__ b2,
                                   unsigned short* __restrict__ y, int M)
{
    int idx = blockIdx.x * blockDim.x + threadIdx.x;
    if (idx >= M * HH) return;
    int m = idx / HH, n = idx % HH;
    float acc = b2[n];
    for (int k = 0; k < BN; ++k) acc += t[m * BN + k] * W2[k * HH + n];
    y[idx] = f32_to_bf16(acc);
}

// ---------------- GRU cell elementwise ----------------
__global__ void gru_cell_kernel(const float* __restrict__ gi, int gi_bstride,
                                const float* __restrict__ gh,
                                float* __restrict__ h,           // in/out (B x H)
                                unsigned short* __restrict__ concat)
{
    int idx = blockIdx.x * blockDim.x + threadIdx.x;
    if (idx >= BB * HH) return;
    int b = idx >> 10, j = idx & (HH - 1);
    const float* gib = gi + (size_t)b * gi_bstride;
    const float* ghb = gh + (size_t)b * H3;
    float ir = gib[j], iz = gib[j + HH], in = gib[j + 2 * HH];
    float hr = ghb[j], hz = ghb[j + HH], hn = ghb[j + 2 * HH];
    float r = 1.0f / (1.0f + __expf(-(ir + hr)));
    float z = 1.0f / (1.0f + __expf(-(iz + hz)));
    float n = tanhf(in + r * hn);
    float hv = (1.0f - z) * n + z * h[idx];
    h[idx] = hv;
    concat[(size_t)b * E4 + j] = f32_to_bf16(hv);
}

// ---------------- attention (f32) ----------------
__global__ __launch_bounds__(256)
void attn_scores_kernel(const float* __restrict__ enc_proj,
                        const float* __restrict__ h,
                        float* __restrict__ w)
{
    int gwave = (blockIdx.x * blockDim.x + threadIdx.x) >> 5;
    int lane  = threadIdx.x & 31;
    if (gwave >= BB * SS) return;
    int b = gwave / SS;
    const float* ep = enc_proj + (size_t)gwave * HH;
    const float* hb = h + (size_t)b * HH;
    float sum = 0.0f;
    for (int e = lane; e < HH; e += 32) sum += ep[e] * hb[e];
#pragma unroll
    for (int off = 16; off; off >>= 1) sum += __shfl_xor(sum, off, 32);
    if (lane == 0) w[gwave] = sum;
}

__global__ __launch_bounds__(128)
void softmax_kernel(const float* __restrict__ w, float* __restrict__ a)
{
    __shared__ float red[128];
    int b = blockIdx.x, t = threadIdx.x;
    float v = (t < SS) ? w[b * SS + t] : -1e30f;
    red[t] = v;
    __syncthreads();
    for (int off = 64; off; off >>= 1) {
        if (t < off) red[t] = fmaxf(red[t], red[t + off]);
        __syncthreads();
    }
    float mx = red[0];
    __syncthreads();
    float e = (t < SS) ? __expf(v - mx) : 0.0f;
    red[t] = e;
    __syncthreads();
    for (int off = 64; off; off >>= 1) {
        if (t < off) red[t] += red[t + off];
        __syncthreads();
    }
    float inv = 1.0f / red[0];
    if (t < SS) a[b * SS + t] = e * inv;
}

// ctx -> bf16 into concat buffer second half
__global__ void ctx_kernel(const float* __restrict__ a,
                           const float* __restrict__ enc_proj,
                           unsigned short* __restrict__ concat)
{
    int idx = blockIdx.x * blockDim.x + threadIdx.x;
    if (idx >= BB * HH) return;
    int b = idx >> 10, j = idx & (HH - 1);
    const float* ab = a + b * SS;
    const float* ep = enc_proj + (size_t)b * SS * HH + j;
    float s = 0.0f;
    for (int t = 0; t < SS; ++t) s += ab[t] * ep[(size_t)t * HH];
    concat[(size_t)b * E4 + HH + j] = f32_to_bf16(s);
}

// ---------------- host orchestration ----------------
extern "C" void kernel_launch(void* const* d_in, const int* in_sizes, int n_in,
                              void* d_out, int out_size, void* d_ws, size_t ws_size,
                              hipStream_t stream) {
    (void)in_sizes; (void)n_in; (void)out_size; (void)ws_size;
    const float* enc_states  = (const float*)d_in[0];
    const float* enc_hidden  = (const float*)d_in[1];
    const float* prev_frames = (const float*)d_in[2];
    const float* ed_W   = (const float*)d_in[3];
    const float* ed_b   = (const float*)d_in[4];
    const float* bn_W1  = (const float*)d_in[5];
    const float* bn_b1  = (const float*)d_in[6];
    const float* bn_W2  = (const float*)d_in[7];
    const float* bn_b2  = (const float*)d_in[8];
    const float* gru_Wih = (const float*)d_in[9];
    const float* gru_Whh = (const float*)d_in[10];
    const float* gru_bih = (const float*)d_in[11];
    const float* gru_bhh = (const float*)d_in[12];
    const float* att_W  = (const float*)d_in[13];
    const float* att_b  = (const float*)d_in[14];
    const float* out_W  = (const float*)d_in[15];
    const float* out_b  = (const float*)d_in[16];
    float* out = (float*)d_out;

    char* p = (char*)d_ws;
    auto alloc = [&](size_t bytes) -> void* {
        void* r = (void*)p;
        p += (bytes + 255) & ~(size_t)255;
        return r;
    };
    unsigned short* ed_Wt    = (unsigned short*)alloc((size_t)HH  * E4 * 2);   // 1024x2048
    unsigned short* gru_Wiht = (unsigned short*)alloc((size_t)H3  * HH * 2);   // 3072x1024
    unsigned short* gru_Whht = (unsigned short*)alloc((size_t)H3  * HH * 2);
    unsigned short* att_Wt   = (unsigned short*)alloc((size_t)HH  * HH * 2);
    unsigned short* out_Wt   = (unsigned short*)alloc((size_t)512 * E4 * 2);   // padded 512x2048
    unsigned short* h0_in    = (unsigned short*)alloc((size_t)BB * E4 * 2);
    unsigned short* enc_bf   = (unsigned short*)alloc((size_t)BB * SS * HH * 2);
    float*          enc_proj = (float*)         alloc((size_t)BB * SS * HH * 4);
    unsigned short* x_hist   = (unsigned short*)alloc((size_t)BB * THIST * HH * 2);
    float*          t_bn     = (float*)         alloc((size_t)BB * THIST * BN * 4);
    float*          gi_hist  = (float*)         alloc((size_t)BB * THIST * H3 * 4);
    float*          gi_dec   = (float*)         alloc((size_t)BB * H3 * 4);
    float*          gh_buf   = (float*)         alloc((size_t)BB * H3 * 4);
    float*          h_f32    = (float*)         alloc((size_t)BB * HH * 4);
    unsigned short* concat   = (unsigned short*)alloc((size_t)BB * E4 * 2);
    unsigned short* x_dec    = (unsigned short*)alloc((size_t)BB * HH * 2);
    float*          wscore   = (float*)         alloc((size_t)BB * SS * 4);
    float*          a_attn   = (float*)         alloc((size_t)BB * SS * 4);

    auto grid1 = [](int n) { return dim3((n + 255) / 256); };
    auto gemm = [&](const unsigned short* A, int lda, const unsigned short* Bt,
                    const float* bias, float* C, int ldc, int M, int N, int K) {
        dim3 g(M / 128, (N + 127) / 128);
        gemm_bf16_kernel<<<g, dim3(256), 0, stream>>>(A, lda, Bt, bias, C, ldc, M, N, K);
    };

    // --- weight conversion (bf16, transposed, padded) ---
    convert_transpose_kernel<<<grid1(HH * E4), 256, 0, stream>>>(ed_W,    ed_Wt,    E4, HH, HH);
    convert_transpose_kernel<<<grid1(H3 * HH), 256, 0, stream>>>(gru_Wih, gru_Wiht, HH, H3, H3);
    convert_transpose_kernel<<<grid1(H3 * HH), 256, 0, stream>>>(gru_Whh, gru_Whht, HH, H3, H3);
    convert_transpose_kernel<<<grid1(HH * HH), 256, 0, stream>>>(att_W,   att_Wt,   HH, HH, HH);
    convert_transpose_kernel<<<grid1(512 * E4), 256, 0, stream>>>(out_W,  out_Wt,   E4, FF, 512);

    // --- activations to bf16 ---
    h0_gather_kernel<<<grid1(BB * E4), 256, 0, stream>>>(enc_hidden, h0_in);
    convert_kernel<<<grid1(BB * SS * HH), 256, 0, stream>>>(enc_states, enc_bf, BB * SS * HH);

    // --- h0 = [enc_hidden] @ ed_W + ed_b ---
    gemm(h0_in, E4, ed_Wt, ed_b, h_f32, HH, BB, HH, E4);
    h_to_concat_kernel<<<grid1(BB * HH), 256, 0, stream>>>(h_f32, concat);

    // --- history bottleneck + batched gi GEMM ---
    bottleneck1_kernel<<<grid1(BB * THIST * BN), 256, 0, stream>>>(
        prev_frames, THIST, 11 * FF, FF, bn_W1, bn_b1, t_bn, BB * THIST);
    bottleneck2_kernel<<<grid1(BB * THIST * HH), 256, 0, stream>>>(
        t_bn, bn_W2, bn_b2, x_hist, BB * THIST);
    gemm(x_hist, HH, gru_Wiht, gru_bih, gi_hist, H3, BB * THIST, H3, HH);

    // --- 10 sequential history GRU steps ---
    for (int t = 0; t < THIST; ++t) {
        gemm(concat, E4, gru_Whht, gru_bhh, gh_buf, H3, BB, H3, HH);
        gru_cell_kernel<<<grid1(BB * HH), 256, 0, stream>>>(
            gi_hist + (size_t)t * H3, THIST * H3, gh_buf, h_f32, concat);
    }

    // --- enc_proj = encoder_states @ att_W + att_b ---
    gemm(enc_bf, HH, att_Wt, att_b, enc_proj, HH, BB * SS, HH, HH);

    // --- 30 decoder steps ---
    const float* frame_base = prev_frames + (size_t)THIST * FF; // prev_frames[:, -1, :]
    int frame_bstride = 11 * FF;
    for (int s = 0; s < DEC; ++s) {
        bottleneck1_kernel<<<grid1(BB * BN), 256, 0, stream>>>(
            frame_base, 1, frame_bstride, 0, bn_W1, bn_b1, t_bn, BB);
        bottleneck2_kernel<<<grid1(BB * HH), 256, 0, stream>>>(
            t_bn, bn_W2, bn_b2, x_dec, BB);
        gemm(x_dec, HH, gru_Wiht, gru_bih, gi_dec, H3, BB, H3, HH);
        gemm(concat, E4, gru_Whht, gru_bhh, gh_buf, H3, BB, H3, HH);
        gru_cell_kernel<<<grid1(BB * HH), 256, 0, stream>>>(
            gi_dec, H3, gh_buf, h_f32, concat);
        attn_scores_kernel<<<dim3(BB * SS * 32 / 256), 256, 0, stream>>>(
            enc_proj, h_f32, wscore);
        softmax_kernel<<<dim3(BB), 128, 0, stream>>>(wscore, a_attn);
        ctx_kernel<<<grid1(BB * HH), 256, 0, stream>>>(a_attn, enc_proj, concat);
        // out_frame written directly into d_out[b, s, :] (ldc = DEC*FF)
        float* C = out + (size_t)s * FF;
        gemm(concat, E4, out_Wt, out_b, C, DEC * FF, BB, FF, E4);
        frame_base = C;
        frame_bstride = DEC * FF;
    }
}
